// MultiHeadAttentionWithRPR_16183436772079
// MI455X (gfx1250) — matrix operand order repaired and checked by reading, not compiled, not run on hardware
//
#include <hip/hip_runtime.h>

// MultiHeadAttentionWithRPR for MI455X (gfx1250), bf16 WMMA path.
// B=4, S=1024, D=1024, H=16, DH=64, CLIP=64.
// Pipeline:
//   1. one-shot fp32->bf16 converts (inputs plain, weights transposed [N][K])
//   2. WMMA GEMM with double-buffered TDM (tensor_load_to_lds) tile staging
//   3. flash-style attention with RPR: async-to-LDS K tiles, WMMA scores/PV,
//      ds_add_f32 rel-pos buckets
//   4. final WMMA GEMM -> fp32 output
// Workspace: Xbf 3*8MB + WT 4*2MB + Q/K/V 3*8MB + concat 8MB + pads ~= 64MB.

constexpr int kB = 4;
constexpr int kS = 1024;
constexpr int kD = 1024;
constexpr int kH = 16;
constexpr int kDH = 64;
constexpr int kCLIP = 64;
constexpr int kR = 129;    // 2*CLIP+1 relative positions
constexpr int kRKP = 144;  // padded to 9 N-tiles of 16
constexpr int kRVP = 160;  // padded to 5 K-steps of 32

typedef __attribute__((ext_vector_type(16))) __bf16 v16bf;
typedef __attribute__((ext_vector_type(8)))  float  v8f;
typedef __attribute__((ext_vector_type(4)))  unsigned int uint4v;
typedef __attribute__((ext_vector_type(8)))  unsigned int uint8v;

union Frag16 {
  uint4v u4[2];
  unsigned int w[8];
  v16bf v;
};

__device__ __forceinline__ unsigned short f2bf(float f) {
  unsigned int x = __float_as_uint(f);
  return (unsigned short)((x + 0x7fffu + ((x >> 16) & 1u)) >> 16);  // RNE
}
__device__ __forceinline__ unsigned int pack2bf(float a, float b) {
  return (unsigned int)f2bf(a) | ((unsigned int)f2bf(b) << 16);
}

__device__ __forceinline__ v8f wmma_bf16(const Frag16& a, const Frag16& b, v8f c) {
  return __builtin_amdgcn_wmma_f32_16x16x32_bf16(false, a.v, false, b.v, (short)0, c,
                                                 false, false);
}

__device__ __forceinline__ unsigned int lds_off(const void* p) {
  return (unsigned int)(size_t)(__attribute__((address_space(3))) const void*)p;
}

// 16-byte global -> LDS async copy (ASYNCcnt path, no VGPR staging).
__device__ __forceinline__ void async_copy16(void* lds, const void* g) {
  unsigned int lo = lds_off(lds);
  asm volatile("global_load_async_to_lds_b128 %0, %1, off" ::"v"(lo), "v"(g)
               : "memory");
}
__device__ __forceinline__ void wait_async0() {
  asm volatile("s_wait_asynccnt 0x0" ::: "memory");
}

// TDM: load a 64-row x 32-element bf16 2D tile (row stride K elements) from
// global into LDS with one tensor_load_to_lds.  D# per CDNA5 ISA §8.3/8.4:
//   group0: count=1 | lds_addr | global_addr | type=2
//   group1: data_size=2B, tensor_dim0=K, tensor_dim1=2^20, tile=32x64,
//           tensor_dim0_stride=K
__device__ __forceinline__ void tdm_load_64x32_bf16(unsigned int lds_addr,
                                                    const unsigned short* g,
                                                    unsigned int K) {
  unsigned long long ga = (unsigned long long)(size_t)g;
  uint4v g0;
  g0[0] = 1u;                                       // count=1, user mode
  g0[1] = lds_addr;                                 // lds_addr[31:0]
  g0[2] = (unsigned int)ga;                         // global_addr[31:0]
  g0[3] = (unsigned int)(ga >> 32) | (2u << 30);    // global_addr[56:32], type=2
  uint8v g1;
  g1[0] = 1u << 16;                                 // data_size=1 (2 bytes)
  g1[1] = (K & 0xFFFFu) << 16;                      // tensor_dim0[15:0]
  g1[2] = (K >> 16);                                // tensor_dim0[31:16], dim1 lo=0
  g1[3] = 16u | (32u << 16);                        // tensor_dim1=2^20, tile_dim0=32
  g1[4] = 64u;                                      // tile_dim1=64, tile_dim2=0
  g1[5] = K;                                        // tensor_dim0_stride[31:0]
  g1[6] = 0u;                                       // stride0 hi, stride1 lo
  g1[7] = 0u;                                       // stride1 hi
  asm volatile("tensor_load_to_lds %0, %1" ::"s"(g0), "s"(g1) : "memory");
}

// ---------------------------------------------------------------------------
// one-shot conversion kernels (bandwidth-bound)
// ---------------------------------------------------------------------------
__global__ void conv_bf16_kernel(const float* __restrict__ in,
                                 unsigned int* __restrict__ out, int n2) {
  int i = blockIdx.x * blockDim.x + threadIdx.x;  // 2 elements per thread
  if (i >= n2) return;
  out[i] = pack2bf(in[2 * i], in[2 * i + 1]);
}

// W[K][N] fp32 -> Wt[N][K] bf16 (transposed, pairs along K)
__global__ void convT_bf16_kernel(const float* __restrict__ W,
                                  unsigned int* __restrict__ Wt, int N, int K) {
  int i = blockIdx.x * blockDim.x + threadIdx.x;  // over N*K/2
  if (i >= N * (K / 2)) return;
  int n = i / (K / 2);
  int kk = (i % (K / 2)) * 2;
  Wt[i] = pack2bf(W[(size_t)kk * N + n], W[(size_t)(kk + 1) * N + n]);
}

__global__ void pad_rprk_kernel(const float* __restrict__ rpr_k,
                                unsigned short* __restrict__ out) {
  int idx = blockIdx.x * blockDim.x + threadIdx.x;  // [kRKP][kDH]
  if (idx >= kRKP * kDH) return;
  int r = idx / kDH, d = idx % kDH;
  out[idx] = (r < kR) ? f2bf(rpr_k[r * kDH + d]) : (unsigned short)0;
}

__global__ void pad_rprvT_kernel(const float* __restrict__ rpr_v,
                                 unsigned short* __restrict__ out) {
  int idx = blockIdx.x * blockDim.x + threadIdx.x;  // [kDH][kRVP]
  if (idx >= kDH * kRVP) return;
  int d = idx / kRVP, r = idx % kRVP;
  out[idx] = (r < kR) ? f2bf(rpr_v[r * kDH + d]) : (unsigned short)0;
}

// ---------------------------------------------------------------------------
// GEMM: C[M,N] = A[M,K] * Bt[N,K]^T.  A, Bt bf16, both row-contiguous.  Tiles
// staged by the Tensor Data Mover (one tensor_load_to_lds per tile, issued by
// wave 0, TENSORcnt synced), double-buffered to overlap DMA with WMMA.
// 128 threads = 4 waves, 64x64 block tile, each wave 32x32 (2x2 fragments).
// ---------------------------------------------------------------------------
__launch_bounds__(128)
__global__ void gemm_bf16wmma(const unsigned short* __restrict__ A,
                              const unsigned short* __restrict__ Bt,
                              float* __restrict__ Cf, unsigned short* __restrict__ Cb,
                              int M, int N, int K, int store_bf) {
  __shared__ unsigned short As[2][64 * 32];  // [m][k]
  __shared__ unsigned short Bs[2][64 * 32];  // [n][k]
  const int tid = threadIdx.x;
  const int lane = tid & 31;
  const int wave = tid >> 5;
  const int bm = blockIdx.y * 64;
  const int bn = blockIdx.x * 64;
  const int wm = (wave >> 1) * 32;
  const int wn = (wave & 1) * 32;
  const int frow = lane & 15;
  const int fsel = lane >> 4;  // which K-half this lane holds
  const int akb = fsel * 8;

  v8f acc[2][2] = {};

  auto issue = [&](int b2, int k0) {
    tdm_load_64x32_bf16(lds_off(&As[b2][0]), &A[(size_t)bm * K + k0], K);
    tdm_load_64x32_bf16(lds_off(&Bs[b2][0]), &Bt[(size_t)bn * K + k0], K);
  };

  if (wave == 0) issue(0, 0);
  int buf = 0;
  for (int k0 = 0; k0 < K; k0 += 32) {
    if (wave == 0) __builtin_amdgcn_s_wait_tensorcnt(0);
    __syncthreads();  // TDM data visible to all waves
    if (wave == 0 && k0 + 32 < K) issue(buf ^ 1, k0 + 32);

    Frag16 fa[2], fb[2];
    for (int i = 0; i < 2; ++i) {
      const unsigned short* ap = &As[buf][(wm + i * 16 + frow) * 32];
      fa[i].u4[0] = *(const uint4v*)(ap + fsel * 16);      // K = fsel*16..+7
      fa[i].u4[1] = *(const uint4v*)(ap + fsel * 16 + 8);  // ..+15
    }
    for (int j = 0; j < 2; ++j) {
      const unsigned short* bp = &Bs[buf][(wn + j * 16 + frow) * 32 + fsel * 16];
      fb[j].u4[0] = *(const uint4v*)(bp);      // K = fsel*16..+7
      fb[j].u4[1] = *(const uint4v*)(bp + 8);  // ..+15
    }
    for (int i = 0; i < 2; ++i)
      for (int j = 0; j < 2; ++j)
        acc[i][j] = wmma_bf16(fa[i], fb[j], acc[i][j]);
    __syncthreads();
    buf ^= 1;
  }

  const int q0 = fsel * 8;
  for (int i = 0; i < 2; ++i)
    for (int j = 0; j < 2; ++j)
      for (int v = 0; v < 8; ++v) {
        size_t gm = (size_t)(bm + wm + i * 16 + q0 + v);
        size_t gn = (size_t)(bn + wn + j * 16 + frow);
        if (store_bf) Cb[gm * N + gn] = f2bf(acc[i][j][v]);
        else          Cf[gm * N + gn] = acc[i][j][v];
      }
}

// ---------------------------------------------------------------------------
// Flash-style attention with RPR.  One wave per block; each block handles a
// 16-query tile of one (b,h).  K tiles streamed into LDS with async-to-LDS
// B128 (double-buffered); online softmax; rel-pos key bias via precomputed
// q.rpr_k^T table (WMMA); rel-pos value bias via probability buckets w[q,r]
// (ds_add_f32) folded in as a final WMMA GEMM.
// ---------------------------------------------------------------------------
__launch_bounds__(32)
__global__ void attn_rpr_kernel(const unsigned short* __restrict__ Qp,
                                const unsigned short* __restrict__ Kp,
                                const unsigned short* __restrict__ Vp,
                                const int* __restrict__ valid_lens,
                                const unsigned short* __restrict__ rprk,   // [kRKP][kDH]
                                const unsigned short* __restrict__ rprvT,  // [kDH][kRVP]
                                unsigned short* __restrict__ concat) {
  const int qt = blockIdx.x;
  const int h = blockIdx.y;
  const int b = blockIdx.z;
  const int lane = threadIdx.x;
  const int vlen = valid_lens[b];
  const int frow = lane & 15;
  const int fsel = lane >> 4;
  const int akb = fsel * 8;
  const int q0 = fsel * 8;  // D-fragment row base for this lane

  __shared__ float qr[16][kRKP];          // q . rpr_k^T bias table
  __shared__ float wbk[16][kRVP];         // probability buckets per rel-pos
  __shared__ float alpha_s[16];
  __shared__ unsigned short pt[16][32];   // probability tile (bf16)
  __shared__ unsigned short vt[64][32];   // V tile transposed [dh][key]
  __shared__ unsigned short Ks[2][32 * 64];  // K tile [key][dh], double-buffered

  for (int i = lane; i < 16 * kRVP; i += 32) (&wbk[0][0])[i] = 0.0f;

  // async-stage a 32-key K tile (rows are 128B contiguous -> 16B chunks)
  auto stage_k = [&](int b2, int kb) {
    for (int c = 0; c < 8; ++c) {
      int idx = c * 32 + lane;  // 256 chunks of 16B
      int row = idx >> 3, c8 = (idx & 7) * 8;
      async_copy16(&Ks[b2][row * 64 + c8],
                   &Kp[((size_t)b * kS + kb + row) * kD + h * kDH + c8]);
    }
  };

  // Q fragments (16 queries x 64 dh = 2 K-steps), kept in registers
  Frag16 qf[2];
  const unsigned short* qrow =
      Qp + ((size_t)b * kS + (size_t)qt * 16 + frow) * kD + h * kDH;
  for (int ks = 0; ks < 2; ++ks) {
    qf[ks].u4[0] = *(const uint4v*)(qrow + ks * 32 + fsel * 16);
    qf[ks].u4[1] = *(const uint4v*)(qrow + ks * 32 + fsel * 16 + 8);
  }

  stage_k(0, 0);

  // qr = Qtile @ rpr_k^T : 9 N-tiles x 2 K-steps
  for (int nt = 0; nt < kRKP / 16; ++nt) {
    v8f acc = {};
    for (int ks = 0; ks < 2; ++ks) {
      Frag16 fb;
      const unsigned short* bp =
          rprk + (size_t)(nt * 16 + frow) * kDH + ks * 32 + fsel * 16;
      fb.u4[0] = *(const uint4v*)bp;
      fb.u4[1] = *(const uint4v*)(bp + 8);
      acc = wmma_bf16(qf[ks], fb, acc);
    }
    for (int v = 0; v < 8; ++v) qr[q0 + v][nt * 16 + frow] = acc[v];
  }
  __syncthreads();

  float m[8], l[8];
  for (int v = 0; v < 8; ++v) { m[v] = -1e30f; l[v] = 0.0f; }
  v8f o[4] = {};

  int kbuf = 0;
  for (int kb = 0; kb < kS; kb += 32) {
    wait_async0();  // K tile for this block is in LDS
    __syncthreads();
    if (kb + 32 < kS) stage_k(kbuf ^ 1, kb + 32);  // overlap next DMA
    // prefetch next V rows
    if (kb + 32 < kS) {
      const unsigned short* vn =
          Vp + ((size_t)b * kS + kb + 32 + lane) * kD + h * kDH;
      __builtin_prefetch(vn, 0, 1);
    }
    // scores for 16q x 32k: two 16-wide N tiles, 2 K-steps each
    v8f sf[2];
    for (int j = 0; j < 2; ++j) {
      v8f acc = {};
      for (int ks = 0; ks < 2; ++ks) {
        Frag16 fb;
        const unsigned short* kp = &Ks[kbuf][(j * 16 + frow) * 64 + ks * 32 + fsel * 16];
        fb.u4[0] = *(const uint4v*)kp;
        fb.u4[1] = *(const uint4v*)(kp + 8);
        acc = wmma_bf16(qf[ks], fb, acc);
      }
      sf[j] = acc;
    }
    // + rpr-key bias, * 1/sqrt(DH), mask, row max
    float mnew[8];
    for (int v = 0; v < 8; ++v) mnew[v] = m[v];
    for (int j = 0; j < 2; ++j) {
      int kg = kb + j * 16 + frow;
      for (int v = 0; v < 8; ++v) {
        int qg = qt * 16 + q0 + v;
        int d = kg - qg;
        d = d < -kCLIP ? -kCLIP : (d > kCLIP ? kCLIP : d);
        float s = (sf[j][v] + qr[q0 + v][d + kCLIP]) * 0.125f;
        if (kg >= vlen) s = -1e6f;
        sf[j][v] = s;
        mnew[v] = fmaxf(mnew[v], s);
      }
    }
    for (int off = 8; off >= 1; off >>= 1)  // reduce within 16-lane halves
      for (int v = 0; v < 8; ++v)
        mnew[v] = fmaxf(mnew[v], __shfl_xor(mnew[v], off, 32));

    float av[8];
    for (int v = 0; v < 8; ++v) {
      av[v] = __expf(m[v] - mnew[v]);
      m[v] = mnew[v];
      l[v] *= av[v];
      for (int nt = 0; nt < 4; ++nt) o[nt][v] *= av[v];
    }
    if (frow == 0)
      for (int v = 0; v < 8; ++v) alpha_s[q0 + v] = av[v];
    __syncthreads();
    for (int i = lane; i < 16 * kRVP; i += 32)
      (&wbk[0][0])[i] *= alpha_s[i / kRVP];
    __syncthreads();

    // probabilities: l, rel-pos buckets, bf16 tile for P.V
    float lsum[8] = {0, 0, 0, 0, 0, 0, 0, 0};
    for (int j = 0; j < 2; ++j) {
      int kg = kb + j * 16 + frow;
      for (int v = 0; v < 8; ++v) {
        float p = __expf(sf[j][v] - m[v]);
        lsum[v] += p;
        int qg = qt * 16 + q0 + v;
        int d = kg - qg;
        d = d < -kCLIP ? -kCLIP : (d > kCLIP ? kCLIP : d);
        atomicAdd(&wbk[q0 + v][d + kCLIP], p);  // ds_add_f32
        pt[q0 + v][j * 16 + frow] = f2bf(p);
      }
    }
    for (int off = 8; off >= 1; off >>= 1)
      for (int v = 0; v < 8; ++v)
        lsum[v] += __shfl_xor(lsum[v], off, 32);
    for (int v = 0; v < 8; ++v) l[v] += lsum[v];

    // stage V tile transposed for B-fragment reads
    for (int i = lane; i < 32 * 64; i += 32) {
      int key = i >> 6, dh = i & 63;
      vt[dh][key] = Vp[((size_t)b * kS + kb + key) * kD + h * kDH + dh];
    }
    __syncthreads();

    // O(16x64) += P(16x32) . V(32x64)
    Frag16 pf;
    const unsigned short* pp = &pt[frow][0];
    pf.u4[0] = *(const uint4v*)(pp + fsel * 16);
    pf.u4[1] = *(const uint4v*)(pp + fsel * 16 + 8);
    for (int nt = 0; nt < 4; ++nt) {
      Frag16 fb;
      const unsigned short* bp = &vt[nt * 16 + frow][fsel * 16];
      fb.u4[0] = *(const uint4v*)bp;
      fb.u4[1] = *(const uint4v*)(bp + 8);
      o[nt] = wmma_bf16(pf, fb, o[nt]);
    }
    __syncthreads();
    kbuf ^= 1;
  }

  // O += w(16 x 160) . rpr_v(160 x 64)   (value-side rel-pos bias)
  for (int ks = 0; ks < kRVP / 32; ++ks) {
    unsigned short tmp[16];
    for (int e = 0; e < 8; ++e) tmp[e] = f2bf(wbk[frow][ks * 32 + fsel * 16 + e]);
    for (int e = 0; e < 8; ++e) tmp[8 + e] = f2bf(wbk[frow][ks * 32 + fsel * 16 + 8 + e]);
    Frag16 wa;
    for (int w = 0; w < 8; ++w)
      wa.w[w] = (unsigned int)tmp[2 * w] | ((unsigned int)tmp[2 * w + 1] << 16);
    for (int nt = 0; nt < 4; ++nt) {
      Frag16 fb;
      const unsigned short* bp =
          rprvT + (size_t)(nt * 16 + frow) * kRVP + ks * 32 + fsel * 16;
      fb.u4[0] = *(const uint4v*)bp;
      fb.u4[1] = *(const uint4v*)(bp + 8);
      o[nt] = wmma_bf16(wa, fb, o[nt]);
    }
  }

  // normalize and write bf16 concat [B,S,D] (head-interleaved)
  for (int nt = 0; nt < 4; ++nt)
    for (int v = 0; v < 8; ++v) {
      int qg = qt * 16 + q0 + v;
      float val = o[nt][v] / l[v];
      concat[((size_t)b * kS + qg) * kD + h * kDH + nt * 16 + frow] = f2bf(val);
    }
}

// ---------------------------------------------------------------------------
extern "C" void kernel_launch(void* const* d_in, const int* in_sizes, int n_in,
                              void* d_out, int out_size, void* d_ws, size_t ws_size,
                              hipStream_t stream) {
  const float* queries = (const float*)d_in[0];
  const float* keys    = (const float*)d_in[1];
  const float* values  = (const float*)d_in[2];
  const int*   vlens   = (const int*)d_in[3];
  const float* Wq = (const float*)d_in[4];
  const float* Wk = (const float*)d_in[5];
  const float* Wv = (const float*)d_in[6];
  const float* Wo = (const float*)d_in[7];
  const float* rpr_k = (const float*)d_in[8];
  const float* rpr_v = (const float*)d_in[9];
  float* out = (float*)d_out;
  (void)in_sizes; (void)n_in; (void)out_size; (void)ws_size;

  const size_t NSD = (size_t)kB * kS * kD;  // 4M elements
  const size_t DD = (size_t)kD * kD;        // 1M elements
  char* ws = (char*)d_ws;
  auto take = [&](size_t bytes) {
    char* p = ws;
    ws += (bytes + 255) & ~(size_t)255;
    return p;
  };
  unsigned short* Xq = (unsigned short*)take(NSD * 2);
  unsigned short* Xk = (unsigned short*)take(NSD * 2);
  unsigned short* Xv = (unsigned short*)take(NSD * 2);
  unsigned short* WqT = (unsigned short*)take(DD * 2);
  unsigned short* WkT = (unsigned short*)take(DD * 2);
  unsigned short* WvT = (unsigned short*)take(DD * 2);
  unsigned short* WoT = (unsigned short*)take(DD * 2);
  unsigned short* Qp = (unsigned short*)take(NSD * 2);
  unsigned short* Kp = (unsigned short*)take(NSD * 2);
  unsigned short* Vp = (unsigned short*)take(NSD * 2);
  unsigned short* concat = (unsigned short*)take(NSD * 2);
  unsigned short* rprk_p  = (unsigned short*)take((size_t)kRKP * kDH * 2);
  unsigned short* rprvT_p = (unsigned short*)take((size_t)kDH * kRVP * 2);

  // one-shot conversions
  const int n2 = (int)(NSD / 2);
  conv_bf16_kernel<<<(n2 + 255) / 256, 256, 0, stream>>>(queries, (unsigned int*)Xq, n2);
  conv_bf16_kernel<<<(n2 + 255) / 256, 256, 0, stream>>>(keys,    (unsigned int*)Xk, n2);
  conv_bf16_kernel<<<(n2 + 255) / 256, 256, 0, stream>>>(values,  (unsigned int*)Xv, n2);
  const int t2 = (int)(DD / 2);
  convT_bf16_kernel<<<(t2 + 255) / 256, 256, 0, stream>>>(Wq, (unsigned int*)WqT, kD, kD);
  convT_bf16_kernel<<<(t2 + 255) / 256, 256, 0, stream>>>(Wk, (unsigned int*)WkT, kD, kD);
  convT_bf16_kernel<<<(t2 + 255) / 256, 256, 0, stream>>>(Wv, (unsigned int*)WvT, kD, kD);
  convT_bf16_kernel<<<(t2 + 255) / 256, 256, 0, stream>>>(Wo, (unsigned int*)WoT, kD, kD);
  pad_rprk_kernel<<<(kRKP * kDH + 255) / 256, 256, 0, stream>>>(rpr_k, rprk_p);
  pad_rprvT_kernel<<<(kDH * kRVP + 255) / 256, 256, 0, stream>>>(rpr_v, rprvT_p);

  // projections (bf16 out)
  dim3 gg(kD / 64, (kB * kS) / 64);
  gemm_bf16wmma<<<gg, 128, 0, stream>>>(Xq, WqT, nullptr, Qp, kB * kS, kD, kD, 1);
  gemm_bf16wmma<<<gg, 128, 0, stream>>>(Xk, WkT, nullptr, Kp, kB * kS, kD, kD, 1);
  gemm_bf16wmma<<<gg, 128, 0, stream>>>(Xv, WvT, nullptr, Vp, kB * kS, kD, kD, 1);

  // attention
  dim3 ga(kS / 16, kH, kB);
  attn_rpr_kernel<<<ga, 32, 0, stream>>>(Qp, Kp, Vp, vlens, rprk_p, rprvT_p, concat);

  // output projection (fp32 out)
  gemm_bf16wmma<<<gg, 128, 0, stream>>>(concat, WoT, out, nullptr, kB * kS, kD, kD, 0);
}
